// RevLSTM_52621939310808
// MI455X (gfx1250) — compile-verified, matrix operand order checked by reading, not run on hardware
//
#include <hip/hip_runtime.h>
#include <math.h>

// CDNA5 wave32 WMMA types
typedef __attribute__((ext_vector_type(16))) __bf16 v16bf;
typedef __attribute__((ext_vector_type(8)))  float  v8f;

// Problem constants: B=4096 rows, IN=1024, g=1024, K=2048, W: 5120 x 2048
#define NROWS      4096
#define GDIM       1024
#define KDIM       2048
#define HIDW       4096            // hidden row width (4*g)
#define OUTH_OFF   16777216        // 4096*4096
#define BITS_OFF   25165824        // + 4096*2048
#define NBLOCKS_L  2048            // 32 x 64 grid per layer

// Fragment-layout workspace byte offsets (total ~92.3 MB)
#define WS_WHI     0u                       // W hi : 5120*2048*2B = 20971520
#define WS_WLO     20971520u                // W lo
#define WS_AHI     41943040u                // A(input cols) hi : 4096*1024*2B
#define WS_ALO     50331648u
#define WS_HHI     58720256u                // A(recurrent cols) hi
#define WS_HLO     67108864u
#define WS_H1FL    75497472u                // h1_fl f32 linear : 16777216 B
#define WS_PART    92274688u                // 4096 doubles

__device__ __forceinline__ float sigf(float x) { return 1.0f / (1.0f + expf(-x)); }

__device__ __forceinline__ void fetch8f(float* dst, const float* __restrict__ src) {
    float4 a = ((const float4*)src)[0];
    float4 b = ((const float4*)src)[1];
    dst[0]=a.x; dst[1]=a.y; dst[2]=a.z; dst[3]=a.w;
    dst[4]=b.x; dst[5]=b.y; dst[6]=b.z; dst[7]=b.w;
}

__device__ __forceinline__ void fetch8i(float* dst, const int* __restrict__ src) {
    int4 a = ((const int4*)src)[0];
    int4 b = ((const int4*)src)[1];
    dst[0]=(float)a.x*0x1p-23f; dst[1]=(float)a.y*0x1p-23f;
    dst[2]=(float)a.z*0x1p-23f; dst[3]=(float)a.w*0x1p-23f;
    dst[4]=(float)b.x*0x1p-23f; dst[5]=(float)b.y*0x1p-23f;
    dst[6]=(float)b.z*0x1p-23f; dst[7]=(float)b.w*0x1p-23f;
}

__device__ __forceinline__ void split16(const float* v, v16bf& hi, v16bf& lo) {
#pragma unroll
    for (int i = 0; i < 16; ++i) {
        float x = v[i];
        __bf16 h = (__bf16)x;
        hi[i] = h;
        lo[i] = (__bf16)(x - (float)h);
    }
}

#define WMMA_BF16(A, B, C) \
    __builtin_amdgcn_wmma_f32_16x16x32_bf16(false, (A), false, (B), (short)0, (C), false, false)

// Load one pre-swizzled bf16 fragment from global: lane-contiguous 32B
__device__ __forceinline__ v16bf ldfrag(const __bf16* __restrict__ base, int frag, int lane) {
    return *(const v16bf*)(base + ((size_t)frag * 32 + lane) * 16);
}

// ---------------------------------------------------------------------------
// Prep: split f32 (or scaled int32) source into hi/lo bf16 fragments stored in
// WMMA per-lane order: dst[frag][lane][16 elems], frag = tile*kTiles + kTile.
// mode 0: W f32 (stride 2048)   mode 1: input f32 (stride 1024)
// mode 2: hidden int32 *2^-23 (stride 4096, col offset 1024)   mode 3: h1fl f32
// ---------------------------------------------------------------------------
__global__ __launch_bounds__(256) void prep_frags(
    const void* __restrict__ src, __bf16* __restrict__ dstHi,
    __bf16* __restrict__ dstLo, int kTiles, int mode)
{
    const int t    = blockIdx.x * blockDim.x + threadIdx.x;
    const int lane = t & 31;
    const int frag = t >> 5;
    const int kT   = frag % kTiles;
    const int tile = frag / kTiles;
    const int row  = tile * 16 + (lane & 15);
    const int half = lane >> 4;
    const int c0   = kT * 32 + 8 * half;

    float v[16];
    if (mode == 0) {
        const float* s = (const float*)src + (size_t)row * KDIM + c0;
        fetch8f(v, s); fetch8f(v + 8, s + 16);
    } else if (mode == 1) {
        const float* s = (const float*)src + (size_t)row * 1024 + c0;
        fetch8f(v, s); fetch8f(v + 8, s + 16);
    } else if (mode == 2) {
        const int* s = (const int*)src + (size_t)row * HIDW + 1024 + c0;
        fetch8i(v, s); fetch8i(v + 8, s + 16);
    } else {
        const float* s = (const float*)src + (size_t)row * 1024 + c0;
        fetch8f(v, s); fetch8f(v + 8, s + 16);
    }

    v16bf hi, lo;
    split16(v, hi, lo);
    *(v16bf*)(dstHi + ((size_t)frag * 32 + lane) * 16) = hi;
    *(v16bf*)(dstLo + ((size_t)frag * 32 + lane) * 16) = lo;
}

// ---------------------------------------------------------------------------
// GEMM with async LDS staging of the block-shared B tile.
// LDS buffer layout per k-step: [wn(2)][q(5)][hl(2)][lane(32)][16 elems] = 20KB.
// Each of the 256 threads issues 5 global_load_async_to_lds_b128 per k-step.
// ---------------------------------------------------------------------------
#define SB_ELEMS 10240   // 20KB / 2B per buffer

__device__ __forceinline__ void issue_b_async(
    const __bf16* __restrict__ wHi, const __bf16* __restrict__ wLo,
    __bf16* sbuf, int kT, int nTile0, int t)
{
#pragma unroll
    for (int i = 0; i < 5; ++i) {
        const int c    = i * 256 + t;          // 16B chunk id, 0..1279
        const int sub  = c & 1;                // low/high half of a 32B fragment-lane
        const int c2   = c >> 1;               // (wn,q,hl,lane) id, 0..639
        const int lnp  = c2 & 31;
        const int rest = c2 >> 5;              // ((wn*5+q)*2+hl), 0..19
        const int wnp  = rest / 10;
        const int r2   = rest % 10;
        const int qq   = r2 >> 1;
        const int hl   = r2 & 1;
        const __bf16* src = (hl ? wLo : wHi)
            + ((size_t)((qq * 64 + nTile0 + wnp) * 64 + kT) * 32 + lnp) * 16 + sub * 8;
        const uint32_t dstLds = (uint32_t)(uintptr_t)(sbuf + (size_t)c * 8);
        const uint64_t ga     = (uint64_t)(uintptr_t)src;
        asm volatile("global_load_async_to_lds_b128 %0, %1, off"
                     :: "v"(dstLds), "v"(ga) : "memory");
    }
}

__global__ __launch_bounds__(256) void rev_lstm_gemm(
    const __bf16* __restrict__ aHi, const __bf16* __restrict__ aLo,
    const __bf16* __restrict__ hHi, const __bf16* __restrict__ hLo,
    const __bf16* __restrict__ wHi, const __bf16* __restrict__ wLo,
    const float* __restrict__ bvec, const int* __restrict__ hid,
    float* __restrict__ out, float* __restrict__ h1fl,
    double* __restrict__ partials, int layer)
{
    __shared__ __align__(16) __bf16 sB[2][SB_ELEMS];   // double-buffered B tile
    __shared__ double sred[256];

    const int t    = threadIdx.x;
    const int lane = t & 31;
    const int wave = t >> 5;
    const int wm   = wave & 3;                    // 4 M-tiles of 16
    const int wn   = wave >> 2;                   // 2 N-tiles of 16
    const int mTile   = blockIdx.y * 4 + wm;      // 0..255
    const int nTile0  = blockIdx.x * 2;           // first of 2 N-tiles in block
    const int nTileLc = nTile0 + wn;              // 0..63 (within one gate group)
    const int rowBase = mTile * 16;
    const int colBase = nTileLc * 16;
    const int nlo  = lane & 15;
    const int half = lane >> 4;

    v8f acc[5] = {};                              // 5 gate groups x 16x16 f32

    // Preload B tile for k-step 0 into buffer 0
    issue_b_async(wHi, wLo, sB[0], 0, nTile0, t);

#pragma unroll 2
    for (int kT = 0; kT < 64; ++kT) {
        const int p = kT & 1;
        // Batch for buffer p (issued last iteration / preloop) must have landed
        // for every wave before anyone reads it; the barrier also guarantees the
        // other buffer is fully consumed before we overwrite it below.
        asm volatile("s_wait_asynccnt 0x0" ::: "memory");
        __syncthreads();
        if (kT + 1 < 64)
            issue_b_async(wHi, wLo, sB[p ^ 1], kT + 1, nTile0, t);  // overlaps WMMAs

        // A fragments: direct coalesced global loads (input half, then recurrent half)
        v16bf ahi, alo;
        if (kT < 32) {
            ahi = ldfrag(aHi, mTile * 32 + kT, lane);
            alo = ldfrag(aLo, mTile * 32 + kT, lane);
        } else {
            ahi = ldfrag(hHi, mTile * 32 + kT - 32, lane);
            alo = ldfrag(hLo, mTile * 32 + kT - 32, lane);
        }

#pragma unroll
        for (int q = 0; q < 5; ++q) {
            const __bf16* bp = &sB[p][(size_t)(((wn * 5 + q) * 2 + 0) * 32 + lane) * 16];
            const v16bf bhi = *(const v16bf*)bp;                 // ds_load_b128 x2
            const v16bf blo = *(const v16bf*)(bp + 512);         // hl=1 plane
            acc[q] = WMMA_BF16(ahi, bhi, acc[q]);
            acc[q] = WMMA_BF16(ahi, blo, acc[q]);
            acc[q] = WMMA_BF16(alo, bhi, acc[q]);
        }
    }

    // ---- Fused epilogue: all 5 gate values for (row,col) are lane-local ----
    const int j = colBase + nlo;
    float bias[5];
#pragma unroll
    for (int q = 0; q < 5; ++q) bias[q] = bvec[q * GDIM + j];

    const int hCol  = (layer == 1 ? 0    : 1024) + j;
    const int cCol  = (layer == 1 ? 2048 : 3072) + j;
    const int ohOff = OUTH_OFF + (layer == 1 ? 0 : 1024);

    double bitsLocal = 0.0;
#pragma unroll
    for (int v = 0; v < 8; ++v) {
        const int row = rowBase + 8 * half + v;
        const float s0 = acc[0][v] + bias[0];
        const float s1 = acc[1][v] + bias[1];
        const float s2 = acc[2][v] + bias[2];
        const float s3 = acc[3][v] + bias[3];
        const float s4 = acc[4][v] + bias[4];

        const float z  = 0.96875f * sigf(s0) + 0.03125f;
        const float gg = tanhf(s1);
        const float f  = sigf(s2);
        const float o  = sigf(s3);
        const float p  = 0.96875f * sigf(s4) + 0.03125f;

        int zq = (int)(z * 1024.0f); zq = zq < 1 ? 1 : zq;
        int pq = (int)(p * 1024.0f); pq = pq < 1 ? 1 : pq;

        const int cOld = hid[(size_t)row * HIDW + cCol];
        const int hOld = hid[(size_t)row * HIDW + hCol];

        const int cNew = (int)(((long long)cOld * (long long)zq) >> 10)
                       + (int)(f * gg * 8388608.0f);
        const float cFl = (float)cNew * 0x1p-23f;
        const int hNew = (int)(((long long)hOld * (long long)pq) >> 10)
                       + (int)(o * tanhf(cFl) * 8388608.0f);
        const float hFl = (float)hNew * 0x1p-23f;

        out[(size_t)row * HIDW + hCol] = (float)hNew;
        out[(size_t)row * HIDW + cCol] = (float)cNew;
        out[(size_t)ohOff + (size_t)row * 2048 + j] = hFl;
        if (layer == 1) h1fl[(size_t)row * 1024 + j] = hFl;

        bitsLocal += (double)(logf(z) + logf(p));
    }

    // Deterministic per-block bits reduction (LDS tree)
    sred[t] = bitsLocal;
    __syncthreads();
#pragma unroll
    for (int s = 128; s > 0; s >>= 1) {
        if (t < s) sred[t] += sred[t + s];
        __syncthreads();
    }
    if (t == 0) {
        partials[(size_t)(layer - 1) * NBLOCKS_L + blockIdx.y * gridDim.x + blockIdx.x] = sred[0];
    }
}

// Fixed-order final reduction -> optimal_bits
__global__ __launch_bounds__(256) void rev_lstm_bits(
    const double* __restrict__ partials, float* __restrict__ bitsOut)
{
    __shared__ double sred[256];
    double s = 0.0;
    for (int i = threadIdx.x; i < 2 * NBLOCKS_L; i += 256) s += partials[i];
    sred[threadIdx.x] = s;
    __syncthreads();
    for (int k = 128; k > 0; k >>= 1) {
        if (threadIdx.x < k) sred[threadIdx.x] += sred[threadIdx.x + k];
        __syncthreads();
    }
    if (threadIdx.x == 0) *bitsOut = (float)(-sred[0] / 0.6931471805599453);
}

extern "C" void kernel_launch(void* const* d_in, const int* in_sizes, int n_in,
                              void* d_out, int out_size, void* d_ws, size_t ws_size,
                              hipStream_t stream) {
    const float* inp = (const float*)d_in[0];   // (4096,1024) f32
    const int*   hid = (const int*)  d_in[1];   // (4096,4096) i32
    const float* W1  = (const float*)d_in[2];   // (5120,2048) f32
    const float* b1  = (const float*)d_in[3];
    const float* W2  = (const float*)d_in[4];
    const float* b2  = (const float*)d_in[5];

    float* out = (float*)d_out;
    char*  ws  = (char*)d_ws;
    __bf16* wHi = (__bf16*)(ws + WS_WHI);
    __bf16* wLo = (__bf16*)(ws + WS_WLO);
    __bf16* aHi = (__bf16*)(ws + WS_AHI);
    __bf16* aLo = (__bf16*)(ws + WS_ALO);
    __bf16* hHi = (__bf16*)(ws + WS_HHI);
    __bf16* hLo = (__bf16*)(ws + WS_HLO);
    float*  h1fl = (float*)(ws + WS_H1FL);
    double* partials = (double*)(ws + WS_PART);

    dim3 block(256);
    dim3 grid(GDIM / 32, NROWS / 64);           // 32 x 64 = 2048 blocks per layer
    const int wBlocks = (320 * 64 * 32) / 256;  // W:  320 nTiles x 64 kTiles
    const int aBlocks = (256 * 32 * 32) / 256;  // A:  256 mTiles x 32 kTiles

    // Layer 1: prep W1, input-A, hidden-A(h2 ints); GEMM+epilogue
    prep_frags<<<wBlocks, block, 0, stream>>>(W1,  wHi, wLo, 64, 0);
    prep_frags<<<aBlocks, block, 0, stream>>>(inp, aHi, aLo, 32, 1);
    prep_frags<<<aBlocks, block, 0, stream>>>(hid, hHi, hLo, 32, 2);
    rev_lstm_gemm<<<grid, block, 0, stream>>>(aHi, aLo, hHi, hLo, wHi, wLo,
                                              b1, hid, out, h1fl, partials, 1);
    // Layer 2: re-prep W buffer with W2, hidden-A from h1_fl; GEMM+epilogue
    prep_frags<<<wBlocks, block, 0, stream>>>(W2,   wHi, wLo, 64, 0);
    prep_frags<<<aBlocks, block, 0, stream>>>(h1fl, hHi, hLo, 32, 3);
    rev_lstm_gemm<<<grid, block, 0, stream>>>(aHi, aLo, hHi, hLo, wHi, wLo,
                                              b2, hid, out, h1fl, partials, 2);

    rev_lstm_bits<<<1, 256, 0, stream>>>(partials, out + BITS_OFF);
}
